// EncoderHead_70325794505322
// MI455X (gfx1250) — matrix operand order repaired and checked by reading, not compiled, not run on hardware
//
#include <hip/hip_runtime.h>
#include <cstdint>
#include <cstddef>

typedef _Float16 v16h __attribute__((ext_vector_type(16)));
typedef _Float16 v8h  __attribute__((ext_vector_type(8)));
typedef float    v8f  __attribute__((ext_vector_type(8)));

#define HID       256
#define NCLS      91
#define NCLSP     96
#define NQ        300
#define BATCH     4
#define S_TOT     21760
#define R_TOT     (BATCH * S_TOT)      // 87040
#define ROWS      32                   // rows per block (2 M-tiles)
#define TILES_PB  (S_TOT / ROWS)       // 680
#define TILES_TOT (R_TOT / ROWS)       // 2720
#define LDA       264                  // padded LDS row stride (halves)
#define CSD       260                  // padded LDS row stride (floats)

#define WMMA_F16(a, b, c) \
  __builtin_amdgcn_wmma_f32_16x16x32_f16(false, (a), false, (b), (short)0, (c), false, false)

__device__ __forceinline__ void level_of(int s, int& lvl, int& p, int& H, int& W) {
  if (s < 256)       { lvl = 0; p = s;        H = 16;  W = 16;  }
  else if (s < 1280) { lvl = 1; p = s - 256;  H = 32;  W = 32;  }
  else if (s < 5376) { lvl = 2; p = s - 1280; H = 64;  W = 64;  }
  else               { lvl = 3; p = s - 5376; H = 128; W = 128; }
}

// A fragment from LDS (row-major, LDA stride f16): two aligned 16B runs per lane.
__device__ __forceinline__ v16h lda_frag(const _Float16* As, int lane, int kbase) {
  int m  = lane & 15;
  int kb = kbase + ((lane & 16) ? 8 : 0);
  const v8h* p = (const v8h*)(As + m * LDA + kb);
  v8h lo = p[0];        // K = kb .. kb+7
  v8h hi = p[2];        // K = kb+16 .. kb+23
  return __builtin_shufflevector(lo, hi, 0, 1, 2, 3, 4, 5, 6, 7,
                                         8, 9, 10, 11, 12, 13, 14, 15);
}

// B fragment from pre-packed weights: one contiguous 32B run per lane.
__device__ __forceinline__ v16h ldb_packed(const _Float16* Wp, int lane, int kk, int nt, int NT) {
  return *(const v16h*)(Wp + (((size_t)(kk * NT + nt) * 32 + lane) << 4));
}

// ---------------- kernel 0: weight packing + proposals ----------------
// dst[((kk*NT + nt)*32 + lane)*16 + j] = W[k][n] (f16),
//   k = kk*32 + ((j&8)?16:0) + ((lane&16)?8:0) + (j&7),  n = nt*16 + (lane&15), zero-padded cols.
__device__ __forceinline__ void pack_weight(const float* src, _Float16* dst,
                                            int NT, int srcCols, int idx, int stride) {
  int total = 8 * NT * 32 * 16;
  for (int i = idx; i < total; i += stride) {
    int j    = i & 15;
    int lane = (i >> 4) & 31;
    int q    = i >> 9;          // kk*NT + nt
    int nt   = q % NT;
    int kk   = q / NT;
    int k = kk * 32 + ((j & 8) ? 16 : 0) + ((lane & 16) ? 8 : 0) + (j & 7);
    int n = nt * 16 + (lane & 15);
    float v = (n < srcCols) ? src[k * srcCols + n] : 0.f;
    dst[i] = (_Float16)v;
  }
}

__global__ void k_prep(const float* projW, const float* encW, const float* clsW,
                       const float* clsB, const float* w1, const float* w2, const float* w3,
                       _Float16* projP, _Float16* encP, _Float16* clsP, float* clsBp,
                       _Float16* w1P, _Float16* w2P, _Float16* w3P,
                       float* props, int* validI) {
  int idx = blockIdx.x * blockDim.x + threadIdx.x;
  int stride = gridDim.x * blockDim.x;
  for (int f = 0; f < 4; ++f)
    pack_weight(projW + (size_t)f * HID * HID, projP + (size_t)f * 8 * 16 * 32 * 16,
                16, HID, idx, stride);
  pack_weight(encW, encP, 16, HID, idx, stride);
  pack_weight(w1, w1P, 16, HID, idx, stride);
  pack_weight(w2, w2P, 16, HID, idx, stride);
  pack_weight(clsW, clsP, 6, NCLS, idx, stride);   // pads n in [91,96)
  pack_weight(w3, w3P, 1, 4, idx, stride);         // pads n in [4,16)
  for (int i = idx; i < NCLSP; i += stride)
    clsBp[i] = (i < NCLS) ? clsB[i] : -3.0e38f;
  for (int s = idx; s < S_TOT; s += stride) {
    int lvl, p, H, W;
    level_of(s, lvl, p, H, W);
    int gi = p / W, gj = p % W;
    float cx = (gj + 0.5f) / (float)W;
    float cy = (gi + 0.5f) / (float)H;
    float wh = 0.05f * (float)(1 << lvl);
    bool valid = (cx > 0.01f) && (cx < 0.99f) && (cy > 0.01f) && (cy < 0.99f) &&
                 (wh > 0.01f) && (wh < 0.99f);
    validI[s] = valid ? 1 : 0;
    float inf = __builtin_inff();
    float lx = logf(cx / (1.f - cx));
    float ly = logf(cy / (1.f - cy));
    float lw = logf(wh / (1.f - wh));
    props[s * 4 + 0] = valid ? lx : inf;
    props[s * 4 + 1] = valid ? ly : inf;
    props[s * 4 + 2] = valid ? lw : inf;
    props[s * 4 + 3] = valid ? lw : inf;
  }
}

// ------------- kernel 1: input-proj GEMM + mask + enc GEMM + LayerNorm -------
__global__ __launch_bounds__(32) void k_proj_enc_ln(
    const float* f0, const float* f1, const float* f2, const float* f3,
    const float* projB, const float* encB, const float* lnG, const float* lnB,
    const _Float16* projP, const _Float16* encP, const int* validI,
    float* memF32, _Float16* memH) {
  __shared__ _Float16 As[ROWS * LDA];
  __shared__ _Float16 Ms[ROWS * LDA];
  __shared__ float    Cs[ROWS * CSD];
  __shared__ float    mu[ROWS], rs[ROWS];

  int lane = threadIdx.x;
  int t = blockIdx.x;
  int b = t / TILES_PB;
  int s0 = (t % TILES_PB) * ROWS;

  int lvl, p0, H, W;
  level_of(s0, lvl, p0, H, W);
  int f = 3 - lvl;
  const float* feat = (f == 0) ? f0 : (f == 1) ? f1 : (f == 2) ? f2 : f3;
  int HW = H * W;

  // Stage A: each lane loads a float4 of 4 contiguous pixels per channel.
  {
    int pq = (lane & 7) * 4;      // pixel quad start (0..28)
    int cq = lane >> 3;           // channel sub-offset (0..3)
    for (int cc = 0; cc < HID; cc += 4) {
      int c = cc + cq;
      const float4 v = *(const float4*)(feat + ((size_t)b * 256 + c) * HW + p0 + pq);
      As[(pq + 0) * LDA + c] = (_Float16)v.x;
      As[(pq + 1) * LDA + c] = (_Float16)v.y;
      As[(pq + 2) * LDA + c] = (_Float16)v.z;
      As[(pq + 3) * LDA + c] = (_Float16)v.w;
    }
  }
  __syncthreads();

  int nlane = lane & 15;
  int mb = (lane & 16) ? 8 : 0;

  int vr[2][8];
#pragma unroll
  for (int mt = 0; mt < 2; ++mt)
#pragma unroll
    for (int i = 0; i < 8; ++i) vr[mt][i] = validI[s0 + mt * 16 + mb + i];

  // --- proj GEMM: (32x256) @ (256x256), masked -> Ms (f16) ---
  const _Float16* Wp = projP + (size_t)f * 8 * 16 * 32 * 16;
  v16h afr[2][8];
  for (int mt = 0; mt < 2; ++mt)
    for (int kk = 0; kk < 8; ++kk) afr[mt][kk] = lda_frag(As + mt * 16 * LDA, lane, kk * 32);
  for (int nt = 0; nt < 16; ++nt) {
    v8f acc0 = {}, acc1 = {};
    for (int kk = 0; kk < 8; ++kk) {
      v16h bfr = ldb_packed(Wp, lane, kk, nt, 16);
      acc0 = WMMA_F16(afr[0][kk], bfr, acc0);
      acc1 = WMMA_F16(afr[1][kk], bfr, acc1);
    }
    float bias = projB[f * HID + nt * 16 + nlane];
#pragma unroll
    for (int i = 0; i < 8; ++i) {
      float x0 = vr[0][i] ? (acc0[i] + bias) : 0.f;
      float x1 = vr[1][i] ? (acc1[i] + bias) : 0.f;
      Ms[(mb + i) * LDA + nt * 16 + nlane] = (_Float16)x0;
      Ms[(16 + mb + i) * LDA + nt * 16 + nlane] = (_Float16)x1;
    }
  }
  __syncthreads();

  // --- enc GEMM: (32x256) @ (256x256) + bias -> Cs (f32) ---
  for (int mt = 0; mt < 2; ++mt)
    for (int kk = 0; kk < 8; ++kk) afr[mt][kk] = lda_frag(Ms + mt * 16 * LDA, lane, kk * 32);
  for (int nt = 0; nt < 16; ++nt) {
    v8f acc0 = {}, acc1 = {};
    for (int kk = 0; kk < 8; ++kk) {
      v16h bfr = ldb_packed(encP, lane, kk, nt, 16);
      acc0 = WMMA_F16(afr[0][kk], bfr, acc0);
      acc1 = WMMA_F16(afr[1][kk], bfr, acc1);
    }
    float bias = encB[nt * 16 + nlane];
#pragma unroll
    for (int i = 0; i < 8; ++i) {
      Cs[(mb + i) * CSD + nt * 16 + nlane] = acc0[i] + bias;
      Cs[(16 + mb + i) * CSD + nt * 16 + nlane] = acc1[i] + bias;
    }
  }
  __syncthreads();

  // --- LayerNorm: one lane per row ---
  {
    int r = lane;
    float s = 0.f;
    for (int c = 0; c < HID; ++c) s += Cs[r * CSD + c];
    float m = s * (1.f / HID);
    float v = 0.f;
    for (int c = 0; c < HID; ++c) { float d = Cs[r * CSD + c] - m; v += d * d; }
    mu[r] = m;
    rs[r] = rsqrtf(v * (1.f / HID) + 1e-5f);
  }
  __syncthreads();

  // --- normalize + write f32 and f16 copies (contiguous 8-wide per lane) ---
  {
    int c0 = lane * 8;
    float g8[8], bb8[8];
#pragma unroll
    for (int jj = 0; jj < 8; ++jj) { g8[jj] = lnG[c0 + jj]; bb8[jj] = lnB[c0 + jj]; }
    size_t rbase = ((size_t)b * S_TOT + s0) * HID;
    for (int i = 0; i < ROWS; ++i) {
      float m = mu[i], r = rs[i];
#pragma unroll
      for (int jj = 0; jj < 8; ++jj) {
        float x = (Cs[i * CSD + c0 + jj] - m) * r * g8[jj] + bb8[jj];
        memF32[rbase + (size_t)i * HID + c0 + jj] = x;
        memH[rbase + (size_t)i * HID + c0 + jj] = (_Float16)x;
      }
    }
  }
}

// ------- kernel 2: cls scores + bbox MLP + coords (shared staged A-tile) ----
__global__ __launch_bounds__(32) void k_heads(
    const _Float16* memH,
    const _Float16* clsP, const float* clsBp,
    const _Float16* w1P, const float* b1,
    const _Float16* w2P, const float* b2,
    const _Float16* w3P, const float* b3,
    const float* props, float* scores, float* coords) {
  __shared__ _Float16 As[ROWS * LDA];
  __shared__ _Float16 Hs[ROWS * LDA];
  __shared__ float    Cs[ROWS * NCLSP];

  int lane = threadIdx.x;
  int t = blockIdx.x;
  int s0 = (t % TILES_PB) * ROWS;
  size_t rbase = (size_t)t * ROWS * HID;

  // Stage mem tile (f16, contiguous 16B per lane per row).
  for (int i = 0; i < ROWS; ++i) {
    const v8h* src = (const v8h*)(memH + rbase + (size_t)i * HID) + lane;
    *(v8h*)(As + i * LDA + lane * 8) = *src;
  }
  __syncthreads();

  int nlane = lane & 15;
  int mb = (lane & 16) ? 8 : 0;
  v16h afr[2][8];
  for (int mt = 0; mt < 2; ++mt)
    for (int kk = 0; kk < 8; ++kk) afr[mt][kk] = lda_frag(As + mt * 16 * LDA, lane, kk * 32);

  // --- cls GEMM (96 padded cols) ---
  for (int nt = 0; nt < 6; ++nt) {
    v8f acc0 = {}, acc1 = {};
    for (int kk = 0; kk < 8; ++kk) {
      v16h bfr = ldb_packed(clsP, lane, kk, nt, 6);
      acc0 = WMMA_F16(afr[0][kk], bfr, acc0);
      acc1 = WMMA_F16(afr[1][kk], bfr, acc1);
    }
    float bias = clsBp[nt * 16 + nlane];
#pragma unroll
    for (int i = 0; i < 8; ++i) {
      Cs[(mb + i) * NCLSP + nt * 16 + nlane] = acc0[i] + bias;
      Cs[(16 + mb + i) * NCLSP + nt * 16 + nlane] = acc1[i] + bias;
    }
  }

  // --- h1 = relu(mem @ w1 + b1) -> Hs (reuses afr) ---
  for (int nt = 0; nt < 16; ++nt) {
    v8f acc0 = {}, acc1 = {};
    for (int kk = 0; kk < 8; ++kk) {
      v16h bfr = ldb_packed(w1P, lane, kk, nt, 16);
      acc0 = WMMA_F16(afr[0][kk], bfr, acc0);
      acc1 = WMMA_F16(afr[1][kk], bfr, acc1);
    }
    float bias = b1[nt * 16 + nlane];
#pragma unroll
    for (int i = 0; i < 8; ++i) {
      Hs[(mb + i) * LDA + nt * 16 + nlane] = (_Float16)fmaxf(acc0[i] + bias, 0.f);
      Hs[(16 + mb + i) * LDA + nt * 16 + nlane] = (_Float16)fmaxf(acc1[i] + bias, 0.f);
    }
  }
  __syncthreads();

  // scores = row-max of cls (one lane per row)
  {
    float mx = -3.4e38f;
    for (int c = 0; c < NCLSP; ++c) mx = fmaxf(mx, Cs[lane * NCLSP + c]);
    scores[t * ROWS + lane] = mx;
  }

  // --- h2 = relu(h1 @ w2 + b2) -> As (safe: reads via regs) ---
  for (int mt = 0; mt < 2; ++mt)
    for (int kk = 0; kk < 8; ++kk) afr[mt][kk] = lda_frag(Hs + mt * 16 * LDA, lane, kk * 32);
  for (int nt = 0; nt < 16; ++nt) {
    v8f acc0 = {}, acc1 = {};
    for (int kk = 0; kk < 8; ++kk) {
      v16h bfr = ldb_packed(w2P, lane, kk, nt, 16);
      acc0 = WMMA_F16(afr[0][kk], bfr, acc0);
      acc1 = WMMA_F16(afr[1][kk], bfr, acc1);
    }
    float bias = b2[nt * 16 + nlane];
#pragma unroll
    for (int i = 0; i < 8; ++i) {
      As[(mb + i) * LDA + nt * 16 + nlane] = (_Float16)fmaxf(acc0[i] + bias, 0.f);
      As[(16 + mb + i) * LDA + nt * 16 + nlane] = (_Float16)fmaxf(acc1[i] + bias, 0.f);
    }
  }
  __syncthreads();

  // --- coords = h2 @ w3pad + b3 + props ---
  for (int mt = 0; mt < 2; ++mt)
    for (int kk = 0; kk < 8; ++kk) afr[mt][kk] = lda_frag(As + mt * 16 * LDA, lane, kk * 32);
  v8f acc0 = {}, acc1 = {};
  for (int kk = 0; kk < 8; ++kk) {
    v16h bfr = ldb_packed(w3P, lane, kk, 0, 1);
    acc0 = WMMA_F16(afr[0][kk], bfr, acc0);
    acc1 = WMMA_F16(afr[1][kk], bfr, acc1);
  }
  if (nlane < 4) {
    float bias = b3[nlane];
#pragma unroll
    for (int i = 0; i < 8; ++i) {
      int r0 = mb + i;
      int r1 = 16 + mb + i;
      coords[((size_t)t * ROWS + r0) * 4 + nlane] = acc0[i] + bias + props[(s0 + r0) * 4 + nlane];
      coords[((size_t)t * ROWS + r1) * 4 + nlane] = acc1[i] + bias + props[(s0 + r1) * 4 + nlane];
    }
  }
}

// ------------- kernel 3: per-batch top-300 (stable) + gather -----------------
__global__ __launch_bounds__(256) void k_topk(const float* memF32, const float* coords,
                                              const float* scores, float* scc_all,
                                              float* out) {
  int b = blockIdx.x;
  int tid = threadIdx.x;
  __shared__ float bs[256];
  __shared__ int   bi[256];

  const float* scb = scores + (size_t)b * S_TOT;
  float* scc = scc_all + (size_t)b * S_TOT;
  for (int i = tid; i < S_TOT; i += 256) scc[i] = scb[i];
  __syncthreads();

  float* tgt = out;                               // (B, NQ, 256)
  float* ref = out + (size_t)BATCH * NQ * HID;    // (B, NQ, 4)

  for (int q = 0; q < NQ; ++q) {
    float best = -__builtin_inff();
    int bidx = 0x7fffffff;
    for (int i = tid; i < S_TOT; i += 256) {
      float v = scc[i];
      if (v > best || (v == best && i < bidx)) { best = v; bidx = i; }
    }
    bs[tid] = best; bi[tid] = bidx;
    __syncthreads();
    for (int st = 128; st > 0; st >>= 1) {
      if (tid < st) {
        float v2 = bs[tid + st]; int i2 = bi[tid + st];
        if (v2 > bs[tid] || (v2 == bs[tid] && i2 < bi[tid])) { bs[tid] = v2; bi[tid] = i2; }
      }
      __syncthreads();
    }
    int s = bi[0];
    if (tid == 0) scc[s] = -__builtin_inff();
    tgt[((size_t)b * NQ + q) * HID + tid] = memF32[((size_t)b * S_TOT + s) * HID + tid];
    if (tid < 4)
      ref[((size_t)b * NQ + q) * 4 + tid] = coords[((size_t)b * S_TOT + s) * 4 + tid];
    __syncthreads();
  }
}

// ------------------------------- launcher -----------------------------------
extern "C" void kernel_launch(void* const* d_in, const int* in_sizes, int n_in,
                              void* d_out, int out_size, void* d_ws, size_t ws_size,
                              hipStream_t stream) {
  (void)in_sizes; (void)n_in; (void)out_size; (void)ws_size;

  const float* feat0 = (const float*)d_in[0];
  const float* feat1 = (const float*)d_in[1];
  const float* feat2 = (const float*)d_in[2];
  const float* feat3 = (const float*)d_in[3];
  const float* projW = (const float*)d_in[4];
  const float* projB = (const float*)d_in[5];
  const float* encW  = (const float*)d_in[6];
  const float* encB  = (const float*)d_in[7];
  const float* lnG   = (const float*)d_in[8];
  const float* lnB   = (const float*)d_in[9];
  const float* clsW  = (const float*)d_in[10];
  const float* clsB  = (const float*)d_in[11];
  const float* bbW1  = (const float*)d_in[12];
  const float* bbB1  = (const float*)d_in[13];
  const float* bbW2  = (const float*)d_in[14];
  const float* bbB2  = (const float*)d_in[15];
  const float* bbW3  = (const float*)d_in[16];
  const float* bbB3  = (const float*)d_in[17];

  char* ws = (char*)d_ws;
  size_t off = 0;
  auto take = [&](size_t bytes) {
    size_t o = off;
    off = (off + bytes + 255) & ~(size_t)255;
    return o;
  };
  size_t o_mem    = take((size_t)R_TOT * HID * 4);
  size_t o_memH   = take((size_t)R_TOT * HID * 2);
  size_t o_scores = take((size_t)R_TOT * 4);
  size_t o_coords = take((size_t)R_TOT * 4 * 4);
  size_t o_props  = take((size_t)S_TOT * 4 * 4);
  size_t o_valid  = take((size_t)S_TOT * 4);
  size_t o_projP  = take((size_t)4 * HID * HID * 2);
  size_t o_encP   = take((size_t)HID * HID * 2);
  size_t o_w1P    = take((size_t)HID * HID * 2);
  size_t o_w2P    = take((size_t)HID * HID * 2);
  size_t o_clsP   = take((size_t)HID * NCLSP * 2);
  size_t o_w3P    = take((size_t)HID * 16 * 2);
  size_t o_clsBp  = take((size_t)NCLSP * 4);
  size_t o_scc    = take((size_t)R_TOT * 4);

  float*     memF32 = (float*)(ws + o_mem);
  _Float16*  memH   = (_Float16*)(ws + o_memH);
  float*     scores = (float*)(ws + o_scores);
  float*     coords = (float*)(ws + o_coords);
  float*     props  = (float*)(ws + o_props);
  int*       validI = (int*)(ws + o_valid);
  _Float16*  projP  = (_Float16*)(ws + o_projP);
  _Float16*  encP   = (_Float16*)(ws + o_encP);
  _Float16*  w1P    = (_Float16*)(ws + o_w1P);
  _Float16*  w2P    = (_Float16*)(ws + o_w2P);
  _Float16*  clsP   = (_Float16*)(ws + o_clsP);
  _Float16*  w3P    = (_Float16*)(ws + o_w3P);
  float*     clsBp  = (float*)(ws + o_clsBp);
  float*     scc    = (float*)(ws + o_scc);

  k_prep<<<256, 256, 0, stream>>>(projW, encW, clsW, clsB, bbW1, bbW2, bbW3,
                                  projP, encP, clsP, clsBp, w1P, w2P, w3P,
                                  props, validI);

  k_proj_enc_ln<<<TILES_TOT, 32, 0, stream>>>(feat0, feat1, feat2, feat3,
                                              projB, encB, lnG, lnB,
                                              projP, encP, validI, memF32, memH);

  k_heads<<<TILES_TOT, 32, 0, stream>>>(memH, clsP, clsBp, w1P, bbB1, w2P, bbB2,
                                        w3P, bbB3, props, scores, coords);

  k_topk<<<BATCH, 256, 0, stream>>>(memF32, coords, scores, scc, (float*)d_out);
}
